// AttentionNMSModule_50680614093490
// MI455X (gfx1250) — compile-verified, matrix operand order checked by reading, not run on hardware
//
#include <hip/hip_runtime.h>
#include <hip/hip_bf16.h>

// ---------------- problem constants ----------------
#define NBQ   300          // boxes
#define NBP   304          // boxes padded to 16
#define NKP   320          // attend K dim padded to 32
#define CLSN  8
#define GRP   16
#define DG    64
#define FEAT  1024
#define EMB   64
#define JTOT  3072         // q_w ++ k_w ++ conv_w rows

typedef __bf16 bf16_t;
typedef bf16_t v16bf __attribute__((ext_vector_type(16)));
typedef float  v8f   __attribute__((ext_vector_type(8)));
typedef float  v4f   __attribute__((ext_vector_type(4)));
typedef unsigned int v4u __attribute__((ext_vector_type(4)));
typedef unsigned int u32x4 __attribute__((ext_vector_type(4)));
typedef int i32x4 __attribute__((ext_vector_type(4)));
typedef int i32x8 __attribute__((ext_vector_type(8)));

// ---------------- CDNA5 feature detection ----------------
#if defined(__has_builtin)
#  if __has_builtin(__builtin_amdgcn_tensor_load_to_lds)
#    define HAVE_TDM 1
#  endif
#  if __has_builtin(__builtin_amdgcn_s_wait_tensorcnt)
#    define WAIT_TENSOR() __builtin_amdgcn_s_wait_tensorcnt(0)
#  else
#    define WAIT_TENSOR() asm volatile("s_wait_tensorcnt 0" ::: "memory")
#  endif
#endif
#ifndef WAIT_TENSOR
#  define WAIT_TENSOR() ((void)0)
#endif
#define COMPILER_MEM_BARRIER() asm volatile("" ::: "memory")

// ---------------- workspace layout (bytes) ----------------
static constexpr size_t XBF_OFF  = 0;                                   // bf16 roi (c,304,1024)
static constexpr size_t XBF_SZ   = (size_t)CLSN*NBP*FEAT*2;
static constexpr size_t WBF_OFF  = XBF_OFF + XBF_SZ;                    // bf16 weights (3072,1024)
static constexpr size_t WBF_SZ   = (size_t)JTOT*FEAT*2;
static constexpr size_t QB_OFF   = WBF_OFF + WBF_SZ;                    // bf16 q (c*g,304,64)
static constexpr size_t QB_SZ    = (size_t)CLSN*GRP*NBP*DG*2;
static constexpr size_t KB_OFF   = QB_OFF + QB_SZ;                      // bf16 k (c*g,304,64)
static constexpr size_t KB_SZ    = QB_SZ;
static constexpr size_t VT_OFF   = KB_OFF + KB_SZ;                      // bf16 V^T (c*g,64,320)
static constexpr size_t VT_SZ    = (size_t)CLSN*GRP*DG*NKP*2;
static constexpr size_t LB_OFF   = VT_OFF + VT_SZ;                      // f32 log-pos (c*g,300,304)
static constexpr size_t LB_SZ    = (size_t)CLSN*GRP*NBQ*NBP*4;
static constexpr size_t PB_OFF   = LB_OFF + LB_SZ;                      // bf16 P (c*g,304,320)
static constexpr size_t PB_SZ    = (size_t)CLSN*GRP*NBP*NKP*2;

// ---------------- helpers ----------------
__device__ __forceinline__ bf16_t f2bf(float f) {
  unsigned u = __builtin_bit_cast(unsigned, f);
  unsigned r = (u + 0x7FFFu + ((u >> 16) & 1u)) >> 16;   // RNE
  unsigned short s = (unsigned short)r;
  return __builtin_bit_cast(bf16_t, s);
}

__device__ __forceinline__ unsigned lds_addr_of(const void* p) {
  // generic shared address: low 32 bits are the LDS byte offset
  return (unsigned)(uintptr_t)p;
}

// WMMA A/B operand load from a row-major bf16 tile (ISA 7.12.2 layout).
__device__ __forceinline__ v16bf load_tile16(const bf16_t* base, int stride,
                                             int lane, int k0) {
  int r  = lane & 15;
  int kh = (lane >> 4) << 3;
  const bf16_t* p = base + (size_t)r * stride + k0 + kh;
  union { v4u u[2]; v16bf v; } x;
  x.u[0] = *(const v4u*)p;          // 8 bf16
  x.u[1] = *(const v4u*)(p + 16);   // 8 bf16
  return x.v;
}

// A operand gather from the TDM-padded bf16 LDS tile:
// row stride 2080B (1024 elems + 2x16B pads), extra 16B pad after elem 512.
__device__ __forceinline__ v16bf load_tileA_lds(const char* s, int lane, int k0) {
  int r  = lane & 15;
  int kh = (lane >> 4) << 3;
  int e0 = k0 + kh;
  int e1 = k0 + 16 + kh;
  const char* row = s + r * 2080;
  union { v4u u[2]; v16bf v; } x;
  x.u[0] = *(const v4u*)(row + e0 * 2 + ((e0 >> 9) << 4));
  x.u[1] = *(const v4u*)(row + e1 * 2 + ((e1 >> 9) << 4));
  return x.v;
}

// f32 -> bf16 operand gather, row-major f32 source (stride in floats).
__device__ __forceinline__ v16bf load_tile16_f32(const float* base, int stride,
                                                 int lane, int k0) {
  int r  = lane & 15;
  int kh = (lane >> 4) << 3;
  const float* p = base + (size_t)r * stride + k0 + kh;
  v4f f0 = *(const v4f*)p;        v4f f1 = *(const v4f*)(p + 4);
  v4f f2 = *(const v4f*)(p + 16); v4f f3 = *(const v4f*)(p + 20);
  v16bf o;
  o[0]=f2bf(f0[0]);  o[1]=f2bf(f0[1]);  o[2]=f2bf(f0[2]);  o[3]=f2bf(f0[3]);
  o[4]=f2bf(f1[0]);  o[5]=f2bf(f1[1]);  o[6]=f2bf(f1[2]);  o[7]=f2bf(f1[3]);
  o[8]=f2bf(f2[0]);  o[9]=f2bf(f2[1]);  o[10]=f2bf(f2[2]); o[11]=f2bf(f2[3]);
  o[12]=f2bf(f3[0]); o[13]=f2bf(f3[1]); o[14]=f2bf(f3[2]); o[15]=f2bf(f3[3]);
  return o;
}

__device__ __forceinline__ v8f wmma_bf16(v8f c, v16bf a, v16bf b) {
  return __builtin_amdgcn_wmma_f32_16x16x32_bf16(false, a, false, b,
                                                 (short)0, c, false, false);
}

// ---------------- TDM 2D tile load: global -> LDS ----------------
// D# group0: count=1, lds_addr, 57-bit global addr, type=2 ("image").
// D# group1: data_size, pad cfg, tensor dims/strides, tile dims.
__device__ __forceinline__ void tdm_load_2d(unsigned lds_off, const void* gaddr,
                                            unsigned dsz_code,       // 0=1B,1=2B,2=4B
                                            unsigned tdim0, unsigned tdim1,
                                            unsigned tile0, unsigned tile1,
                                            unsigned stride0,
                                            unsigned pad_int_code, unsigned pad_amt_code) {
#if defined(HAVE_TDM)
  unsigned long long ga = (unsigned long long)(uintptr_t)gaddr;
  u32x4 g0;
  g0[0] = 1u;                                            // count=1, user desc
  g0[1] = lds_off;                                       // lds_addr
  g0[2] = (unsigned)ga;                                  // global_addr[31:0]
  g0[3] = ((unsigned)(ga >> 32) & 0x1FFFFFFu) | (2u << 30);  // addr[56:32] | type=2
  i32x8 g1;
  g1[0] = (int)((dsz_code << 16) | (1u << 20) |          // data_size, pad_enable
                (pad_int_code << 22) | (pad_amt_code << 25));
  g1[1] = (int)(tdim0 << 16);                            // tensor_dim0[15:0]
  g1[2] = (int)((tdim0 >> 16) | (tdim1 << 16));          // tdim0 hi | tensor_dim1 lo
  g1[3] = (int)((tdim1 >> 16) | (tile0 << 16));          // tdim1 hi | tile_dim0
  g1[4] = (int)tile1;                                    // tile_dim1 (tile_dim2=0)
  g1[5] = (int)stride0;                                  // tensor_dim0_stride[31:0]
  g1[6] = 0;
  g1[7] = 0;                                             // tensor_dim1_stride unused (2D)
  i32x4 z4 = {};
#if __clang_major__ >= 23
  i32x8 z8 = {};
  __builtin_amdgcn_tensor_load_to_lds(g0, g1, z4, z4, z8, 0);
#else
  __builtin_amdgcn_tensor_load_to_lds(g0, g1, z4, z4, 0);
#endif
#else
  (void)lds_off; (void)gaddr; (void)dsz_code; (void)tdim0; (void)tdim1;
  (void)tile0; (void)tile1; (void)stride0; (void)pad_int_code; (void)pad_amt_code;
#endif
}

// ---------------- K0a: pack roi -> bf16, (c, n, f) ----------------
__global__ void pack_x_kernel(const float* __restrict__ roi, bf16_t* __restrict__ xbf) {
  int idx = blockIdx.x * blockDim.x + threadIdx.x;
  const int total = CLSN * NBQ * FEAT;
  if (idx >= total) return;
  int f = idx & (FEAT - 1);
  int t = idx >> 10;
  int n = t % NBQ;
  int c = t / NBQ;
  xbf[((size_t)c * NBP + n) * FEAT + f] =
      f2bf(roi[((size_t)n * CLSN + c) * FEAT + f]);
}

// ---------------- K0b: pack concat(q_w,k_w,conv_w) -> bf16 ----------------
__global__ void pack_w_kernel(const float* __restrict__ qw, const float* __restrict__ kw,
                              const float* __restrict__ cw, bf16_t* __restrict__ wbf) {
  int idx = blockIdx.x * blockDim.x + threadIdx.x;
  const int total = JTOT * FEAT;
  if (idx >= total) return;
  int f = idx & (FEAT - 1);
  int j = idx >> 10;
  float v;
  if (j < 1024)      v = qw[(size_t)j * FEAT + f];
  else if (j < 2048) v = kw[(size_t)(j - 1024) * FEAT + f];
  else               v = cw[(size_t)(j - 2048) * FEAT + f];
  wbf[(size_t)j * FEAT + f] = f2bf(v);
}

// ---------------- K1: fused projection GEMM (2400 x 3072 x 1024) -------------
// A tile (16x1024 bf16) staged once per block via TDM; B streamed from global.
__global__ void proj_kernel(const bf16_t* __restrict__ xbf, const bf16_t* __restrict__ wbf,
                            const float* __restrict__ qb, const float* __restrict__ kb,
                            bf16_t* __restrict__ qbuf, bf16_t* __restrict__ kbuf,
                            bf16_t* __restrict__ vt) {
  __shared__ __align__(16) char sA[16 * 2080];   // padded: +16B per 256 dwords

  const int c  = blockIdx.x / 19;
  const int n0 = (blockIdx.x % 19) * 16;
  const int tid  = threadIdx.x;
  const int wave = tid >> 5;
  const int lane = tid & 31;
  const int jt = blockIdx.y * 8 + wave;       // 0..191
  const int jbase = jt * 16;

  const bf16_t* aT = xbf + ((size_t)c * NBP + n0) * FEAT;
  const bf16_t* bT = wbf + (size_t)jbase * FEAT;

#if defined(HAVE_TDM)
  if (wave == 0) {
    // 2B elems, tensor 1024 x 16, tile 1024 x 16, stride 1024,
    // pad 4 dwords (code 3) every 256 dwords (code 7) -> row stride 2080B.
    tdm_load_2d(lds_addr_of(sA), aT, 1u, FEAT, 16u, FEAT, 16u, FEAT, 7u, 3u);
    WAIT_TENSOR();
    COMPILER_MEM_BARRIER();
  }
#else
  // cooperative copy into the identical padded layout
  for (int t = 0; t < 8; ++t) {
    int ch = t * 256 + tid;                   // 2048 x 16B chunks
    int row = ch >> 7, col16 = ch & 127;
    int seg = col16 >> 6;                     // elem 512 boundary
    v4u v = *(const v4u*)((const char*)aT + (size_t)row * 2048 + col16 * 16);
    *(v4u*)(sA + row * 2080 + col16 * 16 + seg * 16) = v;
  }
#endif
  __syncthreads();

  v8f acc = {};
  #pragma unroll 4
  for (int k0 = 0; k0 < FEAT; k0 += 32) {
    v16bf a = load_tileA_lds(sA, lane, k0);
    v16bf b = load_tile16(bT, FEAT, lane, k0);
    acc = wmma_bf16(acc, a, b);
  }

  const int j = jbase + (lane & 15);
  float bias = 0.0f;
  if (j < 1024)      bias = qb[j];
  else if (j < 2048) bias = kb[j - 1024];
  // conv bias is added in the attend epilogue

  #pragma unroll
  for (int v = 0; v < 8; ++v) {
    int m = ((lane >> 4) << 3) + v;
    int n = n0 + m;
    if (n >= NBQ) continue;
    bf16_t h = f2bf(acc[v] + bias);
    if (j < 1024) {
      int g = j >> 6, d = j & 63;
      qbuf[(((size_t)c * GRP + g) * NBP + n) * DG + d] = h;
    } else if (j < 2048) {
      int j2 = j - 1024, g = j2 >> 6, d = j2 & 63;
      kbuf[(((size_t)c * GRP + g) * NBP + n) * DG + d] = h;
    } else {
      int j3 = j - 2048, g = j3 >> 6, o = j3 & 63;
      vt[(((size_t)c * GRP + g) * DG + o) * NKP + n] = h;
    }
  }
}

// ---------------- K2: pos bias GEMM + relu + log -----------------------------
// L[c,g,q,k] = log(max(relu(pe[c,q,k,:].pos_w[g,:] + pos_b[g]), 1e-6))
// Each wave stages its 16x64 f32 pe tile with one TDM descriptor (OOB rows
// zero-filled by tensor_dim1), padded 4 dwords per 64-dword row in LDS.
__global__ void posbias_kernel(const float* __restrict__ pe, const float* __restrict__ pw,
                               const float* __restrict__ pb, float* __restrict__ lbuf) {
  __shared__ __align__(16) float stage[8][16 * 68];   // 8 waves x 4352B

  const int c = blockIdx.x / NBQ;
  const int q = blockIdx.x % NBQ;
  const int wave = threadIdx.x >> 5;
  const int lane = threadIdx.x & 31;
  const int kt = blockIdx.y * 8 + wave;       // 0..18 used
  if (kt >= 19) return;
  const int k0 = kt * 16;

  const float* aT = pe + (((size_t)c * NBQ + q) * NBQ + k0) * EMB;
  float* sw = &stage[wave][0];
  const unsigned rows_avail = (unsigned)((k0 + 16 <= NBQ) ? 16 : (NBQ - k0));

  // stream hint for the next q's identical tile
  if (q + 1 < NBQ)
    __builtin_prefetch(aT + (size_t)NBQ * EMB, 0, 1);

#if defined(HAVE_TDM)
  // 4B elems, tensor 64 x rows_avail, tile 64 x 16, stride 64,
  // pad 4 dwords (code 3) every 64 dwords (code 5) -> row stride 68 floats.
  tdm_load_2d(lds_addr_of(sw), aT, 2u, EMB, rows_avail, EMB, 16u, EMB, 5u, 3u);
  WAIT_TENSOR();
  COMPILER_MEM_BARRIER();
#else
  for (int t = 0; t < 8; ++t) {
    int ch = t * 32 + lane;                   // 256 x 16B chunks
    int row = ch >> 4, col4 = ch & 15;
    v4f v = {};
    if ((unsigned)row < rows_avail)
      v = *(const v4f*)(aT + (size_t)row * EMB + col4 * 4);
    *(v4f*)(sw + row * 68 + col4 * 4) = v;
  }
#endif

  v8f acc = {};
  #pragma unroll
  for (int d0 = 0; d0 < EMB; d0 += 32) {
    v16bf a = load_tile16_f32(sw, 68, lane, d0);   // LDS gather, conflict-free
    v16bf b = load_tile16_f32(pw, EMB, lane, d0);  // pos_w from global
    acc = wmma_bf16(acc, a, b);
  }

  const int g = lane & 15;
  const float bias = pb[g];
  float* dst = lbuf + (((size_t)c * GRP + g) * NBQ + q) * NBP;
  #pragma unroll
  for (int v = 0; v < 8; ++v) {
    int kk = k0 + ((lane >> 4) << 3) + v;     // < 304
    float x = acc[v] + bias;
    x = fmaxf(x, 0.0f);
    dst[kk] = __logf(fmaxf(x, 1e-6f));
  }
}

// ---------------- K3: aff = q.k^T/8 + L, row softmax, emit P -----------------
__global__ void aff_softmax_kernel(const bf16_t* __restrict__ qbuf,
                                   const bf16_t* __restrict__ kbuf,
                                   const float* __restrict__ lbuf,
                                   float* __restrict__ aff_out,
                                   bf16_t* __restrict__ pbuf) {
  __shared__ float S[16][NBP];
  __shared__ float red[16][8];
  __shared__ float mrow[16];
  __shared__ float sinv[16];

  const int cg = blockIdx.x;                  // c*16+g
  const int q0 = blockIdx.y * 16;
  const int tid = threadIdx.x;                // 128 threads = 4 waves
  const int wave = tid >> 5;
  const int lane = tid & 31;

  const bf16_t* qA = qbuf + ((size_t)cg * NBP + q0) * DG;
  v16bf a0 = load_tile16(qA, DG, lane, 0);
  v16bf a1 = load_tile16(qA, DG, lane, 32);

  for (int kt = wave; kt < 19; kt += 4) {
    const bf16_t* kB = kbuf + ((size_t)cg * NBP + kt * 16) * DG;
    v8f acc = {};
    acc = wmma_bf16(acc, a0, load_tile16(kB, DG, lane, 0));
    acc = wmma_bf16(acc, a1, load_tile16(kB, DG, lane, 32));
    const int kcol = kt * 16 + (lane & 15);
    #pragma unroll
    for (int v = 0; v < 8; ++v) {
      int m = ((lane >> 4) << 3) + v;
      int qr = q0 + m;
      float lv = (qr < NBQ) ? lbuf[((size_t)cg * NBQ + qr) * NBP + kcol] : 0.0f;
      S[m][kcol] = acc[v] * 0.125f + lv;      // 1/sqrt(64)
    }
  }
  __syncthreads();

  // row softmax over k<300 : 8 threads per row
  const int r = tid >> 3, u = tid & 7;
  float mx = -3.0e38f;
  for (int k = u; k < NBQ; k += 8) mx = fmaxf(mx, S[r][k]);
  red[r][u] = mx;
  __syncthreads();
  if (u == 0) {
    float m = red[r][0];
    #pragma unroll
    for (int i = 1; i < 8; ++i) m = fmaxf(m, red[r][i]);
    mrow[r] = m;
  }
  __syncthreads();
  float m = mrow[r], sm = 0.0f;
  for (int k = u; k < NBQ; k += 8) sm += __expf(S[r][k] - m);
  red[r][u] = sm;
  __syncthreads();
  if (u == 0) {
    float s = red[r][0];
    #pragma unroll
    for (int i = 1; i < 8; ++i) s += red[r][i];
    sinv[r] = 1.0f / s;
  }
  __syncthreads();

  for (int idx = tid; idx < 16 * NKP; idx += 128) {
    int rr = idx / NKP, k = idx % NKP;
    int q = q0 + rr;
    if (q >= NBQ) continue;
    float p = (k < NBQ) ? __expf(S[rr][k] - mrow[rr]) * sinv[rr] : 0.0f;
    pbuf[((size_t)cg * NBP + q) * NKP + k] = f2bf(p);
    if (k < NBQ)
      aff_out[((size_t)cg * NBQ + q) * NBQ + k] = p;
  }
}

// ---------------- K4: out = P @ V^T + conv_b  (per c,g: 300x64, K=320) -------
__global__ void attend_kernel(const bf16_t* __restrict__ pbuf,
                              const bf16_t* __restrict__ vt,
                              const float* __restrict__ cb,
                              float* __restrict__ out1) {
  const int cg = blockIdx.x;
  const int q0 = blockIdx.y * 16;
  const int wave = threadIdx.x >> 5;          // 4 waves -> o tiles
  const int lane = threadIdx.x & 31;
  const int o0 = wave * 16;

  const bf16_t* pA = pbuf + ((size_t)cg * NBP + q0) * NKP;
  const bf16_t* vB = vt + ((size_t)cg * DG + o0) * NKP;

  v8f acc = {};
  #pragma unroll
  for (int k0 = 0; k0 < NKP; k0 += 32) {
    v16bf a = load_tile16(pA, NKP, lane, k0);
    v16bf b = load_tile16(vB, NKP, lane, k0);
    acc = wmma_bf16(acc, a, b);
  }

  const int c = cg >> 4, g = cg & 15;
  const int j = g * 64 + o0 + (lane & 15);    // output channel
  const float bias = cb[j];
  #pragma unroll
  for (int v = 0; v < 8; ++v) {
    int q = q0 + ((lane >> 4) << 3) + v;
    if (q >= NBQ) continue;
    out1[((size_t)q * CLSN + c) * (size_t)FEAT + j] = acc[v] + bias;
  }
}

// ---------------- launch -----------------------------------------------------
extern "C" void kernel_launch(void* const* d_in, const int* in_sizes, int n_in,
                              void* d_out, int out_size, void* d_ws, size_t ws_size,
                              hipStream_t stream) {
  const float* roi   = (const float*)d_in[0];
  const float* pe    = (const float*)d_in[1];
  const float* pw    = (const float*)d_in[2];
  const float* pb    = (const float*)d_in[3];
  const float* qw    = (const float*)d_in[4];
  const float* qb    = (const float*)d_in[5];
  const float* kw    = (const float*)d_in[6];
  const float* kb    = (const float*)d_in[7];
  const float* cw    = (const float*)d_in[8];
  const float* cb    = (const float*)d_in[9];

  char* ws = (char*)d_ws;
  bf16_t* xbf  = (bf16_t*)(ws + XBF_OFF);
  bf16_t* wbf  = (bf16_t*)(ws + WBF_OFF);
  bf16_t* qbuf = (bf16_t*)(ws + QB_OFF);
  bf16_t* kbuf = (bf16_t*)(ws + KB_OFF);
  bf16_t* vt   = (bf16_t*)(ws + VT_OFF);
  float*  lbuf = (float*)(ws + LB_OFF);
  bf16_t* pbuf = (bf16_t*)(ws + PB_OFF);

  float* out1 = (float*)d_out;                              // (1,300,8,1024)
  float* aff  = out1 + (size_t)NBQ * CLSN * FEAT;           // (1,128,300,300)

  {
    int total = CLSN * NBQ * FEAT;
    pack_x_kernel<<<(total + 255) / 256, 256, 0, stream>>>(roi, xbf);
  }
  {
    int total = JTOT * FEAT;
    pack_w_kernel<<<(total + 255) / 256, 256, 0, stream>>>(qw, kw, cw, wbf);
  }
  proj_kernel<<<dim3(CLSN * 19, 24), 256, 0, stream>>>(xbf, wbf, qb, kb,
                                                       qbuf, kbuf, vt);
  posbias_kernel<<<dim3(CLSN * NBQ, 3), 256, 0, stream>>>(pe, pw, pb, lbuf);
  aff_softmax_kernel<<<dim3(CLSN * GRP, 19), 128, 0, stream>>>(qbuf, kbuf, lbuf,
                                                               aff, pbuf);
  attend_kernel<<<dim3(CLSN * GRP, 19), 128, 0, stream>>>(pbuf, vt, cb, out1);
  (void)in_sizes; (void)n_in; (void)out_size; (void)ws_size;
}
